// Transformer_36532991820657
// MI455X (gfx1250) — compile-verified, matrix operand order checked by reading, not run on hardware
//
#include <hip/hip_runtime.h>

// ---------------------------------------------------------------------------
// MI455X / gfx1250 transformer forward.  All GEMMs run on the matrix cores
// via v_wmma_f32_16x16x32_bf16 (wave32, 16x16x32 tiles, f32 accumulate).
// LDS tiles are stored K-pair-packed so both A and B fragments are built with
// aligned 32-bit ds loads (compiler merges into ds_load_2addr_b32).
// ---------------------------------------------------------------------------

#define TB 256        // threads per block (8 wave32 waves)
#define BM 128        // block tile M
#define BN 128        // block tile N
#define BK 32         // block tile K (one WMMA K-step)
#define LDK (BK + 2)  // LDS row stride (ushorts) for both tiles, K contiguous

typedef __attribute__((ext_vector_type(16))) __bf16 v16bf;
typedef __attribute__((ext_vector_type(8)))  float  v8f;

union Frag { v16bf v; unsigned u32[8]; };

__device__ __forceinline__ unsigned short f32_to_bf16_rne(float f) {
  union { float f; unsigned u; } c; c.f = f;
  unsigned u = c.u;
  unsigned r = 0x7FFFu + ((u >> 16) & 1u);   // round-to-nearest-even
  return (unsigned short)((u + r) >> 16);
}
__device__ __forceinline__ unsigned pack_bf16x2(float lo, float hi) {
  return (unsigned)f32_to_bf16_rne(lo) | ((unsigned)f32_to_bf16_rne(hi) << 16);
}

// C[z][m][n] = act(alpha * sum_k A[z][m][k] * B[z][k][n] + bias[n])
// Batched with two-level strides: z -> (zb = z/batH, zh = z%batH).
// transB: B element [k][n] read from Bm[n*ldb + k].
// Requirements (met by every call in this model): K % 32 == 0, N % 8 == 0,
// all row pointers 16-byte aligned (lda/ldb multiples of 4).
__global__ __launch_bounds__(TB) void wmma_gemm_kernel(
    const float* __restrict__ A, const float* __restrict__ Bm,
    float* __restrict__ C, const float* __restrict__ bias,
    int M, int N, int K, int lda, int ldb, int ldc,
    float alpha, int relu, int transB, int batH,
    long long sAb, long long sAh, long long sBb, long long sBh,
    long long sCb, long long sCh)
{
  __shared__ unsigned short As[BM * LDK];   // bf16 bits, [m][k] K contiguous
  __shared__ unsigned short Bs[BN * LDK];   // bf16 bits, [n][k] K contiguous

  const int z  = blockIdx.z;
  const int zb = z / batH, zh = z % batH;
  A  += zb * sAb + zh * sAh;
  Bm += zb * sBb + zh * sBh;
  C  += zb * sCb + zh * sCh;

  const int m0 = blockIdx.y * BM;
  const int n0 = blockIdx.x * BN;

  const int tid  = threadIdx.x;
  const int wave = tid >> 5;       // 0..7
  const int lane = tid & 31;
  const int wm   = wave >> 1;      // 4 waves along M (32-row strips)
  const int wn   = wave & 1;       // 2 waves along N (64-col strips)
  const int half = lane >> 4;      // lane group 0-15 / 16-31
  const int l16  = lane & 15;

  v8f acc[2][4] = {};              // 2x4 WMMA tiles per wave (32x64 patch)

  // staging roles
  const int srow = tid >> 1;            // 0..127 (A row / B column)
  const int sseg = (tid & 1) * 16;      // K half-segment 0 / 16
  const int kp   = tid >> 4;            // 0..15  (k-pair, non-trans B)
  const int ng   = tid & 15;            // 0..15  (8-col group, non-trans B)

  for (int k0 = 0; k0 < K; k0 += BK) {
    // ---- stage A tile (BM x BK) -> As[m][k], bf16 pairs packed in dwords ----
    {
      unsigned* dst = (unsigned*)&As[srow * LDK + sseg];   // 8 dwords
      const int gm = m0 + srow;
      if (gm < M) {
        const float4* src =
            (const float4*)(A + (long long)gm * lda + (k0 + sseg));
        if (k0 + BK < K) __builtin_prefetch((const float*)src + BK, 0, 3);
#pragma unroll
        for (int q = 0; q < 4; ++q) {
          const float4 f = src[q];
          dst[2 * q]     = pack_bf16x2(f.x, f.y);
          dst[2 * q + 1] = pack_bf16x2(f.z, f.w);
        }
      } else {
#pragma unroll
        for (int q = 0; q < 8; ++q) dst[q] = 0u;
      }
    }
    // ---- stage B tile (BK x BN) -> Bs[n][k] (column-major, K contiguous) ----
    if (transB) {
      // memory is [n][k]: contiguous K reads, contiguous LDS writes
      unsigned* dst = (unsigned*)&Bs[srow * LDK + sseg];
      const int gn = n0 + srow;
      if (gn < N) {
        const float4* src =
            (const float4*)(Bm + (long long)gn * ldb + (k0 + sseg));
#pragma unroll
        for (int q = 0; q < 4; ++q) {
          const float4 f = src[q];
          dst[2 * q]     = pack_bf16x2(f.x, f.y);
          dst[2 * q + 1] = pack_bf16x2(f.z, f.w);
        }
      } else {
#pragma unroll
        for (int q = 0; q < 8; ++q) dst[q] = 0u;
      }
    } else {
      // memory is [k][n]: read two K rows of 8 columns, transpose into LDS
      const int gk = k0 + 2 * kp;
      const int nb = ng * 8;
      if (n0 + nb + 8 <= N) {
        const float4* r0 = (const float4*)(Bm + (long long)gk * ldb + n0 + nb);
        const float4* r1 =
            (const float4*)(Bm + (long long)(gk + 1) * ldb + n0 + nb);
        const float4 a0 = r0[0], a1 = r0[1];
        const float4 b0 = r1[0], b1 = r1[1];
        const float c0[8] = {a0.x, a0.y, a0.z, a0.w, a1.x, a1.y, a1.z, a1.w};
        const float c1[8] = {b0.x, b0.y, b0.z, b0.w, b1.x, b1.y, b1.z, b1.w};
#pragma unroll
        for (int c = 0; c < 8; ++c)
          *(unsigned*)&Bs[(nb + c) * LDK + 2 * kp] = pack_bf16x2(c0[c], c1[c]);
      } else {
#pragma unroll
        for (int c = 0; c < 8; ++c)
          *(unsigned*)&Bs[(nb + c) * LDK + 2 * kp] = 0u;
      }
    }
    __syncthreads();

    // ---- build fragments (identical dword pattern for A and B) ----
    Frag af[2], bf[4];
#pragma unroll
    for (int i = 0; i < 2; ++i) {
      const int row = wm * 32 + i * 16 + l16;    // A: lane holds row M=l16
#pragma unroll
      for (int r = 0; r < 8; ++r) {
        // VGPR r holds K pair {kk,kk+1}; lanes16-31 shifted by 8
        const int kk = ((r < 4) ? (2 * r) : (16 + 2 * (r - 4))) + half * 8;
        af[i].u32[r] = *(const unsigned*)&As[row * LDK + kk];
      }
    }
#pragma unroll
    for (int j = 0; j < 4; ++j) {
      const int col = wn * 64 + j * 16 + l16;    // B: lane holds column N=l16
#pragma unroll
      for (int r = 0; r < 8; ++r) {
        const int kk = ((r < 4) ? (2 * r) : (16 + 2 * (r - 4))) + half * 8;
        bf[j].u32[r] = *(const unsigned*)&Bs[col * LDK + kk];
      }
    }

#pragma unroll
    for (int i = 0; i < 2; ++i)
#pragma unroll
      for (int j = 0; j < 4; ++j)
        acc[i][j] = __builtin_amdgcn_wmma_f32_16x16x32_bf16(
            false, af[i].v, false, bf[j].v, (short)0, acc[i][j], false, false);
    __syncthreads();
  }

  // ---- epilogue: scale, bias, relu, store (f32 C layout) ----
#pragma unroll
  for (int i = 0; i < 2; ++i) {
#pragma unroll
    for (int j = 0; j < 4; ++j) {
      const int gn = n0 + wn * 64 + j * 16 + l16;
      if (gn >= N) continue;
      const float bv = bias ? bias[gn] : 0.f;
      const int mb = m0 + wm * 32 + i * 16 + half * 8; // VGPR r -> M = r + half*8
#pragma unroll
      for (int r = 0; r < 8; ++r) {
        const int gm = mb + r;
        if (gm < M) {
          float v = alpha * acc[i][j][r] + bv;
          if (relu) v = fmaxf(v, 0.f);
          C[(long long)gm * ldc + gn] = v;
        }
      }
    }
  }
}

// Row softmax over Sk=512 keys with on-the-fly mask.
// mode 0: causal (k > q blocked); mode 1: key padding (tok[k]==0 blocked).
// grid = (Sq, H); scores layout [h][q][k].
__global__ __launch_bounds__(TB) void softmax_mask_kernel(
    float* __restrict__ S, const int* __restrict__ tok, int mode, int Sk)
{
  __shared__ float red[TB];
  const int q = blockIdx.x;
  const long long base = ((long long)blockIdx.y * gridDim.x + q) * Sk;
  const int t = threadIdx.x;
  const int k0 = t, k1 = t + TB;

  float v0 = S[base + k0];
  float v1 = S[base + k1];
  if (mode == 0) {
    if (k0 > q) v0 -= 1e9f;
    if (k1 > q) v1 -= 1e9f;
  } else {
    if (tok[k0] == 0) v0 -= 1e9f;
    if (tok[k1] == 0) v1 -= 1e9f;
  }
  red[t] = fmaxf(v0, v1);
  __syncthreads();
  for (int s = TB / 2; s > 0; s >>= 1) {
    if (t < s) red[t] = fmaxf(red[t], red[t + s]);
    __syncthreads();
  }
  const float m = red[0];
  __syncthreads();
  const float e0 = __expf(v0 - m), e1 = __expf(v1 - m);
  red[t] = e0 + e1;
  __syncthreads();
  for (int s = TB / 2; s > 0; s >>= 1) {
    if (t < s) red[t] += red[t + s];
    __syncthreads();
  }
  const float inv = 1.f / red[0];
  S[base + k0] = e0 * inv;
  S[base + k1] = e1 * inv;
}

// out[row] = LayerNorm(x[row] + a[row]) * g + b  (D = 1024, 4 elems/thread)
__global__ __launch_bounds__(TB) void add_ln_kernel(
    const float* __restrict__ x, const float* __restrict__ a,
    float* __restrict__ out, const float* __restrict__ g,
    const float* __restrict__ bb, int D)
{
  __shared__ float r1[TB], r2[TB];
  const long long row = blockIdx.x;
  const int t = threadIdx.x;
  float v[4], s1 = 0.f, s2 = 0.f;
#pragma unroll
  for (int i = 0; i < 4; ++i) {
    const int d = t + i * TB;
    const float w = x[row * D + d] + a[row * D + d];
    v[i] = w; s1 += w; s2 += w * w;
  }
  r1[t] = s1; r2[t] = s2;
  __syncthreads();
  for (int s = TB / 2; s > 0; s >>= 1) {
    if (t < s) { r1[t] += r1[t + s]; r2[t] += r2[t + s]; }
    __syncthreads();
  }
  const float mean = r1[0] / (float)D;
  const float var  = r2[0] / (float)D - mean * mean;
  const float inv  = rsqrtf(var + 1e-6f);
#pragma unroll
  for (int i = 0; i < 4; ++i) {
    const int d = t + i * TB;
    out[row * D + d] = (v[i] - mean) * inv * g[d] + bb[d];
  }
}

// out[row] = emb[tok[row]] * scale + PE(row % Slen)
__global__ __launch_bounds__(TB) void embed_pe_kernel(
    const int* __restrict__ tok, const float* __restrict__ emb,
    float* __restrict__ out, int Slen, int D, float scale)
{
  const long long row = blockIdx.x;
  const int pos = (int)(row % Slen);
  const int tk = tok[row];
  const float ln10k = 9.210340371976184f;  // ln(10000)
#pragma unroll
  for (int i = 0; i < 4; ++i) {
    const int d = threadIdx.x + i * TB;
    const float expo = (float)(2 * (d >> 1)) / (float)D;
    const float ang = (float)pos * __expf(-expo * ln10k);
    const float pe = (d & 1) ? __cosf(ang) : __sinf(ang);
    out[row * D + d] = emb[(long long)tk * D + d] * scale + pe;
  }
}

// ---------------------------------------------------------------------------
// Host orchestration
// ---------------------------------------------------------------------------
static void gemm(hipStream_t st, const float* A, const float* B, float* C,
                 const float* bias, int M, int N, int K, int lda, int ldb,
                 int ldc, float alpha, int relu, int transB,
                 int batch = 1, int batH = 1,
                 long long sAb = 0, long long sAh = 0,
                 long long sBb = 0, long long sBh = 0,
                 long long sCb = 0, long long sCh = 0)
{
  dim3 grid((N + BN - 1) / BN, (M + BM - 1) / BM, batch);
  wmma_gemm_kernel<<<grid, TB, 0, st>>>(A, B, C, bias, M, N, K, lda, ldb, ldc,
                                        alpha, relu, transB, batH,
                                        sAb, sAh, sBb, sBh, sCb, sCh);
}

// Multi-head attention: T = MHA(Xq; Xkv) with weights W[4,D,D], bias4[4,D].
// maskmode: 0 = causal, 1 = key padding from tok.
static void run_attention(hipStream_t st, const float* Xq, const float* Xkv,
                          const float* W, const float* bias4,
                          float* Qb, float* Kb, float* Vb, float* Sc,
                          float* Ab, float* T, const int* tok, int maskmode)
{
  const int Bz = 8, S = 512, D = 1024, Hh = 16, Dh = 64;
  const long long SD = (long long)S * D, SS = (long long)S * S,
                  DD = (long long)D * D;
  gemm(st, Xq,  W + 0 * DD, Qb, bias4 + 0 * D, Bz * S, D, D, D, D, D, 1.f, 0, 0);
  gemm(st, Xkv, W + 1 * DD, Kb, bias4 + 1 * D, Bz * S, D, D, D, D, D, 1.f, 0, 0);
  gemm(st, Xkv, W + 2 * DD, Vb, bias4 + 2 * D, Bz * S, D, D, D, D, D, 1.f, 0, 0);
  for (int b = 0; b < Bz; ++b) {
    // scores[h] = (Q_h @ K_h^T) / 8   (batched over heads)
    gemm(st, Qb + b * SD, Kb + b * SD, Sc, nullptr,
         S, S, Dh, D, D, S, 0.125f, 0, /*transB=*/1,
         Hh, Hh, 0, Dh, 0, Dh, 0, SS);
    softmax_mask_kernel<<<dim3(S, Hh), TB, 0, st>>>(
        Sc, tok ? tok + b * S : nullptr, maskmode, S);
    // context[h] = P_h @ V_h  -> concatenated heads in Ab
    gemm(st, Sc, Vb + b * SD, Ab + b * SD, nullptr,
         S, Dh, S, S, D, D, 1.f, 0, 0,
         Hh, Hh, 0, SS, 0, Dh, 0, Dh);
  }
  gemm(st, Ab, W + 3 * DD, T, bias4 + 3 * D, Bz * S, D, D, D, D, D, 1.f, 0, 0);
}

extern "C" void kernel_launch(void* const* d_in, const int* in_sizes, int n_in,
                              void* d_out, int out_size, void* d_ws,
                              size_t ws_size, hipStream_t stream)
{
  (void)in_sizes; (void)n_in; (void)out_size; (void)ws_size;
  const int Bz = 8, S = 512, D = 1024, DF = 4096, L = 6, VT = 32000;
  const long long SD  = (long long)S * D;
  const long long BSD = (long long)Bz * SD;

  const int*   enc_in  = (const int*)  d_in[0];
  const int*   dec_in  = (const int*)  d_in[1];
  const float* src_emb = (const float*)d_in[2];
  const float* tgt_emb = (const float*)d_in[3];
  const float* e_aw    = (const float*)d_in[4];
  const float* e_ab    = (const float*)d_in[5];
  const float* e_f1w   = (const float*)d_in[6];
  const float* e_f1b   = (const float*)d_in[7];
  const float* e_f2w   = (const float*)d_in[8];
  const float* e_f2b   = (const float*)d_in[9];
  const float* e_lng   = (const float*)d_in[10];
  const float* e_lnb   = (const float*)d_in[11];
  const float* d_aw    = (const float*)d_in[12];
  const float* d_ab    = (const float*)d_in[13];
  const float* d_f1w   = (const float*)d_in[14];
  const float* d_f1b   = (const float*)d_in[15];
  const float* d_f2w   = (const float*)d_in[16];
  const float* d_f2b   = (const float*)d_in[17];
  const float* d_lng   = (const float*)d_in[18];
  const float* d_lnb   = (const float*)d_in[19];
  const float* out_w   = (const float*)d_in[20];
  const float* out_b   = (const float*)d_in[21];
  float* logits = (float*)d_out;

  // Workspace layout (floats): 7 state/temp buffers of B*S*D, then a 64MB
  // region shared by per-batch attention scores (16 MB) and FFN hidden.
  float* ws = (float*)d_ws;
  float* X  = ws;            // encoder state  [B,S,D]
  float* Y  = X  + BSD;      // decoder state  [B,S,D]
  float* Qb = Y  + BSD;
  float* Kb = Qb + BSD;
  float* Vb = Kb + BSD;
  float* Ab = Vb + BSD;      // attention context (heads concat)
  float* T  = Ab + BSD;      // sublayer output pre-LN
  float* Sc = T  + BSD;      // scores [H,S,S] per batch / FFN hidden [B*S,DF]

  // ---------------- encoder ----------------
  embed_pe_kernel<<<dim3(Bz * S), TB, 0, stream>>>(enc_in, src_emb, X, S, D, 32.0f);
  for (int l = 0; l < L; ++l) {
    const float* W  = e_aw + (long long)l * 4 * D * D;
    const float* bb = e_ab + (long long)l * 4 * D;
    run_attention(stream, X, X, W, bb, Qb, Kb, Vb, Sc, Ab, T, enc_in, 1);
    add_ln_kernel<<<dim3(Bz * S), TB, 0, stream>>>(
        X, T, X, e_lng + (2 * l + 0) * D, e_lnb + (2 * l + 0) * D, D);
    gemm(stream, X, e_f1w + (long long)l * D * DF, Sc, e_f1b + (long long)l * DF,
         Bz * S, DF, D, D, DF, DF, 1.f, /*relu=*/1, 0);
    gemm(stream, Sc, e_f2w + (long long)l * DF * D, T, e_f2b + (long long)l * D,
         Bz * S, D, DF, DF, D, D, 1.f, 0, 0);
    add_ln_kernel<<<dim3(Bz * S), TB, 0, stream>>>(
        X, T, X, e_lng + (2 * l + 1) * D, e_lnb + (2 * l + 1) * D, D);
  }

  // ---------------- decoder ----------------
  embed_pe_kernel<<<dim3(Bz * S), TB, 0, stream>>>(dec_in, tgt_emb, Y, S, D, 32.0f);
  for (int l = 0; l < L; ++l) {
    const float* W0 = d_aw + (long long)(2 * l + 0) * 4 * D * D;
    const float* b0 = d_ab + (long long)(2 * l + 0) * 4 * D;
    run_attention(stream, Y, Y, W0, b0, Qb, Kb, Vb, Sc, Ab, T, nullptr, 0);
    add_ln_kernel<<<dim3(Bz * S), TB, 0, stream>>>(
        Y, T, Y, d_lng + (3 * l + 0) * D, d_lnb + (3 * l + 0) * D, D);

    const float* W1 = d_aw + (long long)(2 * l + 1) * 4 * D * D;
    const float* b1 = d_ab + (long long)(2 * l + 1) * 4 * D;
    run_attention(stream, Y, X, W1, b1, Qb, Kb, Vb, Sc, Ab, T, enc_in, 1);
    add_ln_kernel<<<dim3(Bz * S), TB, 0, stream>>>(
        Y, T, Y, d_lng + (3 * l + 1) * D, d_lnb + (3 * l + 1) * D, D);

    gemm(stream, Y, d_f1w + (long long)l * D * DF, Sc, d_f1b + (long long)l * DF,
         Bz * S, DF, D, D, DF, DF, 1.f, 1, 0);
    gemm(stream, Sc, d_f2w + (long long)l * DF * D, T, d_f2b + (long long)l * D,
         Bz * S, D, DF, DF, D, D, 1.f, 0, 0);
    add_ln_kernel<<<dim3(Bz * S), TB, 0, stream>>>(
        Y, T, Y, d_lng + (3 * l + 2) * D, d_lnb + (3 * l + 2) * D, D);
  }

  // ---------------- output projection ----------------
  gemm(stream, Y, out_w, logits, out_b, Bz * S, VT, D, D, VT, VT, 1.f, 0, 0);
}
